// SpatialGraphConv_88742614270232
// MI455X (gfx1250) — matrix operand order repaired and checked by reading, not compiled, not run on hardware
//
#include <hip/hip_runtime.h>
#include <hip/hip_bf16.h>

#define NROWS 8192
#define CDIM  128
#define KSEG  2048          // K-split: 4 segments of 2048
#define LDS_STRIDE 48       // 32 + 16 pad: rows 96B (16B aligned), bank spread
#define WSTR 264            // 256 + 8 pad: rows 528B (16B aligned), 16 lanes -> 64 banks

typedef __attribute__((ext_vector_type(16))) __bf16 v16bf;
typedef __attribute__((ext_vector_type(8)))  __bf16 v8bf;
typedef __attribute__((ext_vector_type(8)))  float  v8f;

__device__ __forceinline__ __bf16 f2bf(float f) {
  unsigned int u = __builtin_bit_cast(unsigned int, f);
  u += 0x7fffu + ((u >> 16) & 1u);        // round-to-nearest-even
  unsigned short h = (unsigned short)(u >> 16);
  return __builtin_bit_cast(__bf16, h);
}
__device__ __forceinline__ float bf2f(__bf16 b) {
  unsigned int u = ((unsigned int)__builtin_bit_cast(unsigned short, b)) << 16;
  return __builtin_bit_cast(float, u);
}
// Pack two floats into two RNE-rounded bf16 in one dword (lo -> bits 15:0).
__device__ __forceinline__ unsigned int pack2bf(float lo, float hi) {
  unsigned int ul = __builtin_bit_cast(unsigned int, lo);
  ul += 0x7fffu + ((ul >> 16) & 1u);
  unsigned int uh = __builtin_bit_cast(unsigned int, hi);
  uh += 0x7fffu + ((uh >> 16) & 1u);
  return (ul >> 16) | (uh & 0xffff0000u);
}

__global__ void zero_kernel(float* __restrict__ p, int n) {
  int i = blockIdx.x * blockDim.x + threadIdx.x;
  if (i < n) p[i] = 0.0f;
}

// ---------------------------------------------------------------------------
// Kernel 2: S += A^T @ X over this block's K segment; degree row-partials
// fused so A is read from HBM exactly once total.
// grid = (64 M-blocks, 4 K-segments), block = 256 (8 waves)
// ---------------------------------------------------------------------------
__global__ void __launch_bounds__(256)
aggregate_wmma_kernel(const float* __restrict__ A,
                      const float* __restrict__ X,
                      float* __restrict__ S,
                      float* __restrict__ deg) {
  __shared__ __bf16 Alds[128 * LDS_STRIDE];   // [i_local][j_local]  (A^T tile)
  __shared__ __bf16 Xlds[128 * LDS_STRIDE];   // [c][j_local]        (B tile)

  const int tid  = threadIdx.x;
  const int w    = tid >> 5;
  const int L    = tid & 31;
  const int half = (L < 16) ? 0 : 1;
  const int iBase    = blockIdx.x * 128;
  const int kSegBase = blockIdx.y * KSEG;

  v8f acc[8] = {};
  const int aoff0 = (16 * w + (L & 15)) * LDS_STRIDE + half * 8;

  for (int it = 0; it < KSEG / 32; ++it) {
    const int k0 = kSegBase + it * 32;
    __syncthreads();
    // Stage 32(j) x 128(i/c) fp32 -> transposed bf16 LDS tiles.
    // Unit = (j-pair, i-group-of-4): float4 loads (coalesced across threads),
    // two adjacent-K bf16 packed per ds_store_b32.
    #pragma unroll
    for (int u0 = 0; u0 < 2; ++u0) {
      const int u  = tid + u0 * 256;        // 0..511
      const int jp = u >> 5;                // 0..15  -> j = 2*jp
      const int ig = u & 31;                // 0..31  -> i = 4*ig
      const float* pa = &A[(size_t)(k0 + 2 * jp) * NROWS + iBase + 4 * ig];
      float4 a0 = *(const float4*)pa;
      float4 a1 = *(const float4*)(pa + NROWS);
      *(unsigned int*)&Alds[(4 * ig + 0) * LDS_STRIDE + 2 * jp] = pack2bf(a0.x, a1.x);
      *(unsigned int*)&Alds[(4 * ig + 1) * LDS_STRIDE + 2 * jp] = pack2bf(a0.y, a1.y);
      *(unsigned int*)&Alds[(4 * ig + 2) * LDS_STRIDE + 2 * jp] = pack2bf(a0.z, a1.z);
      *(unsigned int*)&Alds[(4 * ig + 3) * LDS_STRIDE + 2 * jp] = pack2bf(a0.w, a1.w);
      const float* px = &X[(size_t)(k0 + 2 * jp) * CDIM + 4 * ig];
      float4 x0 = *(const float4*)px;
      float4 x1 = *(const float4*)(px + CDIM);
      *(unsigned int*)&Xlds[(4 * ig + 0) * LDS_STRIDE + 2 * jp] = pack2bf(x0.x, x1.x);
      *(unsigned int*)&Xlds[(4 * ig + 1) * LDS_STRIDE + 2 * jp] = pack2bf(x0.y, x1.y);
      *(unsigned int*)&Xlds[(4 * ig + 2) * LDS_STRIDE + 2 * jp] = pack2bf(x0.z, x1.z);
      *(unsigned int*)&Xlds[(4 * ig + 3) * LDS_STRIDE + 2 * jp] = pack2bf(x0.w, x1.w);
    }
    __syncthreads();

    // Degree partials for rows j = k0..k0+31 (A elements are exact 0/1 in bf16).
    if (tid < 32) {
      float s = 0.0f;
      #pragma unroll 16
      for (int i = 0; i < 128; ++i) s += bf2f(Alds[i * LDS_STRIDE + tid]);
      atomicAdd(&deg[k0 + tid], s);
    }

    // Prefetch next A tile (global_prefetch_b8).
    if (it + 1 < KSEG / 32) {
      __builtin_prefetch(&A[(size_t)(k0 + 32 + (tid >> 3)) * NROWS + iBase + (tid & 7) * 16], 0, 1);
    }

    // A fragment (16x32 bf16): lane<16 -> K 0-7 & 16-23, lane>=16 -> K 8-15 & 24-31.
    v8bf a0 = *(const v8bf*)&Alds[aoff0];
    v8bf a1 = *(const v8bf*)&Alds[aoff0 + 16];
    v16bf a;
    #pragma unroll
    for (int i = 0; i < 8; ++i) { a[i] = a0[i]; a[i + 8] = a1[i]; }

    // Preload all 8 B fragments (distinct registers) so ds_loads can run
    // ahead of the WMMA chain instead of draining dscnt before each WMMA.
    v16bf bfr[8];
    #pragma unroll
    for (int t = 0; t < 8; ++t) {
      int boff = (t * 16 + (L & 15)) * LDS_STRIDE + half * 8;  // lane<16: K0-7 & 16-23, else K8-15 & 24-31
      v8bf b0 = *(const v8bf*)&Xlds[boff];
      v8bf b1 = *(const v8bf*)&Xlds[boff + 16];
      #pragma unroll
      for (int i = 0; i < 8; ++i) { bfr[t][i] = b0[i]; bfr[t][i + 8] = b1[i]; }
    }
    #pragma unroll
    for (int t = 0; t < 8; ++t) {
      acc[t] = __builtin_amdgcn_wmma_f32_16x16x32_bf16(
                 false, a, false, bfr[t], (short)0, acc[t], false, false);
    }
  }

  // Merge K-split partials. C/D layout: VGPR r -> row r + 8*half, col = L&15.
  #pragma unroll
  for (int t = 0; t < 8; ++t) {
    const int col = t * 16 + (L & 15);
    #pragma unroll
    for (int r = 0; r < 8; ++r) {
      int row = iBase + 16 * w + r + half * 8;
      atomicAdd(&S[row * CDIM + col], acc[t][r]);
    }
  }
}

// ---------------------------------------------------------------------------
// Kernel 3: h = [deg^-1 * S | X] @ [W_n | W_c]^T + (b_n + b_c), one K=256 WMMA
// GEMM, plus per-channel sum / sum-of-squares for BatchNorm batch stats.
// grid = 64, block = 256, dynamic LDS.
// ---------------------------------------------------------------------------
__global__ void __launch_bounds__(256)
fused_linear_stats_kernel(const float* __restrict__ S,
                          const float* __restrict__ X,
                          const float* __restrict__ Wn,
                          const float* __restrict__ bn,
                          const float* __restrict__ Wc,
                          const float* __restrict__ bc,
                          const float* __restrict__ deg,
                          float* __restrict__ h,
                          float* __restrict__ gSum,
                          float* __restrict__ gSq) {
  extern __shared__ __bf16 smem[];
  __bf16* Wlds = smem;                              // [c][k'] 128 x WSTR
  float* chSum = (float*)(smem + 128 * WSTR);       // 128
  float* chSq  = chSum + 128;                       // 128

  const int tid  = threadIdx.x;
  const int w    = tid >> 5;
  const int L    = tid & 31;
  const int half = (L < 16) ? 0 : 1;
  const int iBase = blockIdx.x * 128;

  // Stage combined weights (vectorized): k'<128 -> W_n[c][k'], else W_c.
  #pragma unroll 8
  for (int r = 0; r < 32; ++r) {
    int idx4 = tid + r * 256;                 // 0..8191 float4 units
    int c = idx4 >> 6, k = (idx4 & 63) * 4;   // 64 float4 per 256-wide row
    float4 v = (k < 128) ? *(const float4*)&Wn[c * 128 + k]
                         : *(const float4*)&Wc[c * 128 + (k - 128)];
    *(unsigned int*)&Wlds[c * WSTR + k]     = pack2bf(v.x, v.y);
    *(unsigned int*)&Wlds[c * WSTR + k + 2] = pack2bf(v.z, v.w);
  }
  if (tid < 128) { chSum[tid] = 0.0f; chSq[tid] = 0.0f; }
  __syncthreads();

  const int row = iBase + 16 * w + (L & 15);   // A-operand row for this lane
  float dv = deg[row];
  float scale = (dv == 0.0f) ? 1.0f : 1.0f / dv;

  v8f acc[8] = {};
  #pragma unroll
  for (int kc = 0; kc < 8; ++kc) {
    const float* src = (kc < 4) ? &S[row * CDIM + kc * 32]
                                : &X[row * CDIM + (kc - 4) * 32];
    const float sc = (kc < 4) ? scale : 1.0f;
    const int o0 = half * 8;
    float fl[16];
    *(float4*)(&fl[0])  = *(const float4*)(src + o0);
    *(float4*)(&fl[4])  = *(const float4*)(src + o0 + 4);
    *(float4*)(&fl[8])  = *(const float4*)(src + o0 + 16);
    *(float4*)(&fl[12]) = *(const float4*)(src + o0 + 20);
    v16bf a;
    #pragma unroll
    for (int i = 0; i < 16; ++i) a[i] = f2bf(fl[i] * sc);

    v16bf bfr[8];
    #pragma unroll
    for (int t = 0; t < 8; ++t) {
      int boff = (t * 16 + (L & 15)) * WSTR + kc * 32 + half * 8;
      v8bf b0 = *(const v8bf*)&Wlds[boff];
      v8bf b1 = *(const v8bf*)&Wlds[boff + 16];
      #pragma unroll
      for (int i = 0; i < 8; ++i) { bfr[t][i] = b0[i]; bfr[t][i + 8] = b1[i]; }
    }
    #pragma unroll
    for (int t = 0; t < 8; ++t) {
      acc[t] = __builtin_amdgcn_wmma_f32_16x16x32_bf16(
                 false, a, false, bfr[t], (short)0, acc[t], false, false);
    }
  }

  // Bias + store h + per-channel batch statistics.
  #pragma unroll
  for (int t = 0; t < 8; ++t) {
    const int col = t * 16 + (L & 15);
    const float bsum = bn[col] + bc[col];
    float s1 = 0.0f, s2 = 0.0f;
    #pragma unroll
    for (int r = 0; r < 8; ++r) {
      int orow = iBase + 16 * w + r + half * 8;
      float v = acc[t][r] + bsum;
      h[orow * CDIM + col] = v;
      s1 += v; s2 += v * v;
    }
    s1 += __shfl_xor(s1, 16);   // combine lane pairs sharing a column
    s2 += __shfl_xor(s2, 16);
    if (L < 16) { atomicAdd(&chSum[col], s1); atomicAdd(&chSq[col], s2); }
  }
  __syncthreads();
  if (tid < 128) { atomicAdd(&gSum[tid], chSum[tid]); atomicAdd(&gSq[tid], chSq[tid]); }
}

// ---------------------------------------------------------------------------
// Kernel 4: BatchNorm (training-mode batch stats, biased var) + affine + ReLU
// ---------------------------------------------------------------------------
__global__ void bn_relu_kernel(const float* __restrict__ h,
                               const float* __restrict__ gSum,
                               const float* __restrict__ gSq,
                               const float* __restrict__ gamma,
                               const float* __restrict__ beta,
                               float* __restrict__ out) {
  int i = blockIdx.x * blockDim.x + threadIdx.x;
  if (i >= NROWS * CDIM) return;
  int c = i & (CDIM - 1);
  const float invN = 1.0f / (float)NROWS;
  float mu  = gSum[c] * invN;
  float var = gSq[c] * invN - mu * mu;
  float rs  = rsqrtf(var + 1e-5f);
  float y   = gamma[c] * (h[i] - mu) * rs + beta[c];
  out[i] = fmaxf(y, 0.0f);
}

extern "C" void kernel_launch(void* const* d_in, const int* in_sizes, int n_in,
                              void* d_out, int out_size, void* d_ws, size_t ws_size,
                              hipStream_t stream) {
  (void)in_sizes; (void)n_in; (void)out_size; (void)ws_size;
  const float* X     = (const float*)d_in[0];   // node_features [8192,128]
  const float* A     = (const float*)d_in[1];   // adjacency    [8192,8192]
  const float* Wn    = (const float*)d_in[2];
  const float* bn    = (const float*)d_in[3];
  const float* Wc    = (const float*)d_in[4];
  const float* bc    = (const float*)d_in[5];
  const float* gamma = (const float*)d_in[6];
  const float* beta  = (const float*)d_in[7];

  float* ws   = (float*)d_ws;
  float* S    = ws;                       // 8192*128 = 1048576 floats
  float* deg  = S + NROWS * CDIM;         // 8192
  float* gSum = deg + NROWS;              // 128
  float* gSq  = gSum + CDIM;              // 128
  float* h    = gSq + CDIM;               // 8192*128

  const int zn = NROWS * CDIM + NROWS + 2 * CDIM;   // S + deg + sums
  zero_kernel<<<(zn + 255) / 256, 256, 0, stream>>>(ws, zn);

  aggregate_wmma_kernel<<<dim3(NROWS / 128, NROWS / KSEG), 256, 0, stream>>>(A, X, S, deg);

  size_t smem = (size_t)128 * WSTR * sizeof(__bf16) + 2 * CDIM * sizeof(float);
  fused_linear_stats_kernel<<<NROWS / 128, 256, smem, stream>>>(
      S, X, Wn, bn, Wc, bc, deg, h, gSum, gSq);

  bn_relu_kernel<<<(NROWS * CDIM + 255) / 256, 256, 0, stream>>>(
      h, gSum, gSq, gamma, beta, (float*)d_out);
}
